// AGDNConv_14173392077040
// MI455X (gfx1250) — matrix operand order repaired and checked by reading, not compiled, not run on hardware
//
#include <hip/hip_runtime.h>
#include <hip/hip_bf16.h>
#include <math.h>

// Problem constants (match reference)
#define NN 50000
#define EE 800000
#define IND 256      // input feature dim
#define HD 256       // H*D
#define HH 4
#define DD 64
#define KK 3
#define NEG_SLOPE 0.2f

typedef __attribute__((ext_vector_type(16))) __bf16 bf16x16;
typedef __attribute__((ext_vector_type(8)))  __bf16 bf16x8;
typedef __attribute__((ext_vector_type(8)))  float  f32x8;
typedef __attribute__((ext_vector_type(4)))  int    v4i;

union ABFrag { bf16x16 v; bf16x8 h[2]; };

// CDNA5 async global->LDS copies (ASYNCcnt) if the toolchain exposes them.
// Probe-confirmed signature: (AS1 int4*, AS3 int4*, imm offset, imm cpol).
#if defined(__gfx1250__) && __has_builtin(__builtin_amdgcn_global_load_async_to_lds_b128) && \
    __has_builtin(__builtin_amdgcn_s_wait_asynccnt)
#define HAVE_ASYNC_LDS 1
#define ASYNC_B128(g, l)                                                     \
    __builtin_amdgcn_global_load_async_to_lds_b128(                          \
        (__attribute__((address_space(1))) v4i*)(g),                         \
        (__attribute__((address_space(3))) v4i*)(l), 0, 0)
#else
#define HAVE_ASYNC_LDS 0
#endif

__device__ __forceinline__ float leaky(float x) {
    return x >= 0.f ? x : NEG_SLOPE * x;
}

__device__ __forceinline__ float wave_reduce_sum(float v) {
#pragma unroll
    for (int off = 16; off > 0; off >>= 1) v += __shfl_xor(v, off, 32);
    return v;
}

// float atomic-max via int/uint punning (works for mixed signs, init = -inf)
__device__ __forceinline__ void atomicMaxF(float* addr, float val) {
    if (val >= 0.f) atomicMax((int*)addr, __float_as_int(val));
    else            atomicMin((unsigned int*)addr, __float_as_uint(val));
}

// ---------------------------------------------------------------------------
// fill / convert
// ---------------------------------------------------------------------------
__global__ void fill_f32(float* __restrict__ p, long long count, float v) {
    long long i = (long long)blockIdx.x * blockDim.x + threadIdx.x;
    long long stride = (long long)gridDim.x * blockDim.x;
    for (; i < count; i += stride) p[i] = v;
}

__global__ void cvt_f32_bf16(const float* __restrict__ in, __bf16* __restrict__ out,
                             long long count) {
    long long i = (long long)blockIdx.x * blockDim.x + threadIdx.x;
    long long stride = (long long)gridDim.x * blockDim.x;
    for (; i < count; i += stride) out[i] = (__bf16)in[i];
}

// ---------------------------------------------------------------------------
// Kernel 1: feat_src = feat @ W^T via v_wmma_f32_16x16x32_bf16.
// Block: 128 threads (4 waves), 64x64 output tile, BK=32, K=256.
// Inputs pre-packed to bf16 (featb [NN x 256], Wb [256 x 256]).
// Tiles staged to LDS with async global->LDS b128 copies (double-buffered,
// next tile's copies overlap current tile's WMMAs); sync uint4 fallback.
// ---------------------------------------------------------------------------
#define LDA 40   // 32 + 8 bf16 pad
#define LDB 40

__global__ void __launch_bounds__(128)
gemm_feat_wmma(const __bf16* __restrict__ featb, const __bf16* __restrict__ Wb,
               float* __restrict__ h0) {
    __shared__ __align__(16) __bf16 As[2][64 * LDA];
    __shared__ __align__(16) __bf16 Bs[2][64 * LDB];

    const int tid  = threadIdx.x;
    const int lane = tid & 31;
    const int wv   = tid >> 5;          // wave 0..3
    const int m0   = blockIdx.x * 64;   // row tile
    const int n0   = blockIdx.y * 64;   // col tile (0,64,128,192)

    f32x8 acc[4] = {};

    auto compute = [&](int buf) {
        const int mrow  = wv * 16 + (lane & 15);
        const int khalf = lane >> 4;
        ABFrag af;
        af.h[0] = *(const bf16x8*)&As[buf][mrow * LDA + khalf * 8];
        af.h[1] = *(const bf16x8*)&As[buf][mrow * LDA + 16 + khalf * 8];
#pragma unroll
        for (int ns = 0; ns < 4; ++ns) {
            const int nrow = ns * 16 + (lane & 15);
            ABFrag bfr;
            bfr.h[0] = *(const bf16x8*)&Bs[buf][nrow * LDB + khalf * 8];
            bfr.h[1] = *(const bf16x8*)&Bs[buf][nrow * LDB + 16 + khalf * 8];
            acc[ns] = __builtin_amdgcn_wmma_f32_16x16x32_bf16(
                false, af.v, false, bfr.v, (short)0, acc[ns], false, false);
        }
    };

#if HAVE_ASYNC_LDS
    // one 16-byte chunk per thread per issue; 2 issues cover a 64x32 bf16 tile
    auto stage = [&](int buf, int k0) {
#pragma unroll
        for (int i = 0; i < 2; ++i) {
            int c = i * 128 + tid;                 // chunk 0..255
            int row = c >> 2, col8 = (c & 3) * 8;
            int gm = m0 + row; if (gm > NN - 1) gm = NN - 1;   // clamp, no branch
            ASYNC_B128(&featb[(size_t)gm * IND + k0 + col8],
                       &As[buf][row * LDA + col8]);
            ASYNC_B128(&Wb[(size_t)(n0 + row) * IND + k0 + col8],
                       &Bs[buf][row * LDB + col8]);
        }
    };
    stage(0, 0);
    // steady state: stage next buffer, wait for current 4 asyncs, compute.
#define GEMM_STEP(ks)                                        \
    {   stage(((ks) + 1) & 1, ((ks) + 1) * 32);              \
        __builtin_amdgcn_s_wait_asynccnt(4);                 \
        __syncthreads();                                     \
        compute((ks) & 1);                                   \
        __syncthreads(); }
    GEMM_STEP(0) GEMM_STEP(1) GEMM_STEP(2) GEMM_STEP(3)
    GEMM_STEP(4) GEMM_STEP(5) GEMM_STEP(6)
#undef GEMM_STEP
    // drain
    __builtin_amdgcn_s_wait_asynccnt(0);
    __syncthreads();
    compute(1);
#else
    for (int k0 = 0; k0 < IND; k0 += 32) {
#pragma unroll
        for (int i = 0; i < 2; ++i) {
            int c = i * 128 + tid;
            int row = c >> 2, col8 = (c & 3) * 8;
            int gm = m0 + row; if (gm > NN - 1) gm = NN - 1;
            *(uint4*)&As[0][row * LDA + col8] =
                *(const uint4*)&featb[(size_t)gm * IND + k0 + col8];
            *(uint4*)&Bs[0][row * LDB + col8] =
                *(const uint4*)&Wb[(size_t)(n0 + row) * IND + k0 + col8];
        }
        __syncthreads();
        compute(0);
        __syncthreads();
    }
#endif

    // store C/D: VGPR r -> M = r + 8*(lane/16), N = lane%16
#pragma unroll
    for (int ns = 0; ns < 4; ++ns) {
        int ncol = n0 + ns * 16 + (lane & 15);
#pragma unroll
        for (int r = 0; r < 8; ++r) {
            int grow = m0 + wv * 16 + r + 8 * (lane >> 4);
            if (grow < NN) h0[(size_t)grow * HD + ncol] = acc[ns][r];
        }
    }
}

// ---------------------------------------------------------------------------
// Kernel 2: el[n,h] = <h0[n,h,:], attn_l[h,:]>, er likewise. One wave / node.
// ---------------------------------------------------------------------------
__global__ void el_er_kernel(const float* __restrict__ h0,
                             const float* __restrict__ attn_l,
                             const float* __restrict__ attn_r,
                             float* __restrict__ el, float* __restrict__ er) {
    int lane = threadIdx.x & 31;
    int n = (blockIdx.x * blockDim.x + threadIdx.x) >> 5;
    if (n >= NN) return;
    const float* row = &h0[(size_t)n * HD];
    float pl[4] = {0, 0, 0, 0}, pr[4] = {0, 0, 0, 0};
#pragma unroll
    for (int j = 0; j < 8; ++j) {
        int i = lane + 32 * j;          // head = j>>1 (compile-time)
        float v = row[i];
        pl[j >> 1] += v * attn_l[i];
        pr[j >> 1] += v * attn_r[i];
    }
#pragma unroll
    for (int h = 0; h < 4; ++h) {
        float rl = wave_reduce_sum(pl[h]);
        float rr = wave_reduce_sum(pr[h]);
        if (lane == 0) { el[n * HH + h] = rl; er[n * HH + h] = rr; }
    }
}

// ---------------------------------------------------------------------------
// Kernel 3: per-edge logits + segment max (atomic float max into m[dst,h])
// ---------------------------------------------------------------------------
__global__ void edge_logit_max(const float* __restrict__ el, const float* __restrict__ er,
                               const int* __restrict__ src, const int* __restrict__ dst,
                               float* __restrict__ abuf, float* __restrict__ mbuf) {
    int e = blockIdx.x * blockDim.x + threadIdx.x;
    if (e >= EE) return;
    int s = src[e], d = dst[e];
    float4 l4 = ((const float4*)el)[s];
    float4 r4 = ((const float4*)er)[d];
    float4 v;
    v.x = leaky(l4.x + r4.x); v.y = leaky(l4.y + r4.y);
    v.z = leaky(l4.z + r4.z); v.w = leaky(l4.w + r4.w);
    ((float4*)abuf)[e] = v;
    atomicMaxF(&mbuf[d * HH + 0], v.x);
    atomicMaxF(&mbuf[d * HH + 1], v.y);
    atomicMaxF(&mbuf[d * HH + 2], v.z);
    atomicMaxF(&mbuf[d * HH + 3], v.w);
}

// ---------------------------------------------------------------------------
// Kernel 4: ex = exp(logit - m[dst]); segment sum into s[dst,h]
// ---------------------------------------------------------------------------
__global__ void edge_exp_sum(const int* __restrict__ dst,
                             float* __restrict__ abuf,
                             const float* __restrict__ mbuf,
                             float* __restrict__ sbuf) {
    int e = blockIdx.x * blockDim.x + threadIdx.x;
    if (e >= EE) return;
    int d = dst[e];
    float4 lg = ((const float4*)abuf)[e];
    float4 m4 = ((const float4*)mbuf)[d];
    float4 ex;
    ex.x = __expf(lg.x - m4.x); ex.y = __expf(lg.y - m4.y);
    ex.z = __expf(lg.z - m4.z); ex.w = __expf(lg.w - m4.w);
    ((float4*)abuf)[e] = ex;
    atomicAdd(&sbuf[d * HH + 0], ex.x);
    atomicAdd(&sbuf[d * HH + 1], ex.y);
    atomicAdd(&sbuf[d * HH + 2], ex.z);
    atomicAdd(&sbuf[d * HH + 3], ex.w);
}

// ---------------------------------------------------------------------------
// Kernel 5: a = ex / s[dst]
// ---------------------------------------------------------------------------
__global__ void edge_norm(const int* __restrict__ dst, float* __restrict__ abuf,
                          const float* __restrict__ sbuf) {
    int e = blockIdx.x * blockDim.x + threadIdx.x;
    if (e >= EE) return;
    int d = dst[e];
    float4 ex = ((const float4*)abuf)[e];
    float4 s4 = ((const float4*)sbuf)[d];
    float4 a;
    a.x = s4.x > 0.f ? ex.x / s4.x : 0.f;
    a.y = s4.y > 0.f ? ex.y / s4.y : 0.f;
    a.z = s4.z > 0.f ? ex.z / s4.z : 0.f;
    a.w = s4.w > 0.f ? ex.w / s4.w : 0.f;
    ((float4*)abuf)[e] = a;
}

// ---------------------------------------------------------------------------
// Kernel 6: one diffusion hop. One wave per edge (grid-stride);
// lane covers 8 contiguous floats of the 256-float row -> coalesced float4
// gathers + contiguous global_atomic_add_f32. Prefetch next source row.
// ---------------------------------------------------------------------------
__global__ void spmm_hop(const float* __restrict__ hprev,
                         const float* __restrict__ abuf,
                         const int* __restrict__ src, const int* __restrict__ dst,
                         float* __restrict__ hnext) {
    int lane = threadIdx.x & 31;
    int wid  = (blockIdx.x * blockDim.x + threadIdx.x) >> 5;
    int nw   = (gridDim.x * blockDim.x) >> 5;
    for (int e = wid; e < EE; e += nw) {
        int ep = e + nw;
        if (ep < EE && lane == 0)
            __builtin_prefetch(&hprev[(size_t)src[ep] * HD], 0, 0);  // global_prefetch_b8
        int s = src[e], d = dst[e];
        float w = abuf[e * HH + (lane >> 3)];     // head = (lane*8)/64
        const float4* hp = (const float4*)&hprev[(size_t)s * HD];
        float4 x0 = hp[lane * 2 + 0];
        float4 x1 = hp[lane * 2 + 1];
        float* out = &hnext[(size_t)d * HD + lane * 8];
        atomicAdd(out + 0, w * x0.x); atomicAdd(out + 1, w * x0.y);
        atomicAdd(out + 2, w * x0.z); atomicAdd(out + 3, w * x0.w);
        atomicAdd(out + 4, w * x1.x); atomicAdd(out + 5, w * x1.y);
        atomicAdd(out + 6, w * x1.z); atomicAdd(out + 7, w * x1.w);
    }
}

// ---------------------------------------------------------------------------
// Kernel 7: hop attention + output. One wave per node. pos_emb added on the
// fly; per-(node,head) softmax over K+1=4 hops; + bias.
// ---------------------------------------------------------------------------
__global__ void hop_attn_kernel(const float* __restrict__ h0, const float* __restrict__ h1,
                                const float* __restrict__ h2, const float* __restrict__ h3,
                                const float* __restrict__ hop_l, const float* __restrict__ hop_r,
                                const float* __restrict__ pos_emb, const float* __restrict__ bias,
                                float* __restrict__ out) {
    int lane = threadIdx.x & 31;
    int n = (blockIdx.x * blockDim.x + threadIdx.x) >> 5;
    if (n >= NN) return;
    const float* rows[4] = { &h0[(size_t)n * HD], &h1[(size_t)n * HD],
                             &h2[(size_t)n * HD], &h3[(size_t)n * HD] };
    float hv[4][8];
    float al[4][4] = {{0}};       // [hop][head]
    float ar[4] = {0, 0, 0, 0};   // [head], from hop 0
#pragma unroll
    for (int k = 0; k < 4; ++k) {
        const float* row = rows[k];
#pragma unroll
        for (int j = 0; j < 8; ++j) {
            int i = lane + 32 * j;
            int head = j >> 1;            // compile-time per unrolled j
            int d = i & 63;
            float v = row[i] + pos_emb[head * ((KK + 1) * DD) + k * DD + d];
            hv[k][j] = v;
            al[k][head] += v * hop_l[i];
            if (k == 0) ar[head] += v * hop_r[i];
        }
    }
    float AL[4][4], AR[4];
#pragma unroll
    for (int k = 0; k < 4; ++k)
#pragma unroll
        for (int h = 0; h < 4; ++h) AL[k][h] = wave_reduce_sum(al[k][h]);
#pragma unroll
    for (int h = 0; h < 4; ++h) AR[h] = wave_reduce_sum(ar[h]);

    float wgt[4][4];                      // [head][hop]
#pragma unroll
    for (int h = 0; h < 4; ++h) {
        float lg[4], mx = -INFINITY;
#pragma unroll
        for (int k = 0; k < 4; ++k) { lg[k] = leaky(AL[k][h] + AR[h]); mx = fmaxf(mx, lg[k]); }
        float s = 0.f;
#pragma unroll
        for (int k = 0; k < 4; ++k) { lg[k] = __expf(lg[k] - mx); s += lg[k]; }
#pragma unroll
        for (int k = 0; k < 4; ++k) wgt[h][k] = lg[k] / s;
    }
#pragma unroll
    for (int j = 0; j < 8; ++j) {
        int i = lane + 32 * j;
        int head = j >> 1;
        float acc = bias[i];
#pragma unroll
        for (int k = 0; k < 4; ++k) acc += wgt[head][k] * hv[k][j];
        out[(size_t)n * HD + i] = acc;
    }
}

// ---------------------------------------------------------------------------
extern "C" void kernel_launch(void* const* d_in, const int* in_sizes, int n_in,
                              void* d_out, int out_size, void* d_ws, size_t ws_size,
                              hipStream_t stream) {
    const float* feat   = (const float*)d_in[0];
    const int*   src    = (const int*)d_in[1];
    const int*   dst    = (const int*)d_in[2];
    const float* W_src  = (const float*)d_in[3];
    const float* attn_l = (const float*)d_in[4];
    const float* attn_r = (const float*)d_in[5];
    const float* hop_l  = (const float*)d_in[6];
    const float* hop_r  = (const float*)d_in[7];
    const float* pos    = (const float*)d_in[8];
    const float* bias   = (const float*)d_in[9];
    float* out = (float*)d_out;

    // Workspace layout (~221 MB). featb aliases h3: consumed by the GEMM,
    // h3 is zero-filled AFTER the GEMM and before hop 3 writes it.
    const size_t NF = (size_t)NN * HD;
    float* h0   = (float*)d_ws;
    float* h1   = h0 + NF;
    float* h2   = h1 + NF;
    float* h3   = h2 + NF;
    float* el   = h3 + NF;
    float* er   = el + (size_t)NN * HH;
    float* mbuf = er + (size_t)NN * HH;
    float* sbuf = mbuf + (size_t)NN * HH;
    float* abuf = sbuf + (size_t)NN * HH;
    __bf16* featb = (__bf16*)h3;                       // NF bf16 (fits in h3)
    __bf16* Wb    = (__bf16*)(abuf + (size_t)EE * HH); // 64K bf16

    // init: zero h1,h2; m = -inf; s = 0
    fill_f32<<<2048, 256, 0, stream>>>(h1, (long long)2 * NF, 0.f);
    fill_f32<<<256, 256, 0, stream>>>(mbuf, (long long)NN * HH, -INFINITY);
    fill_f32<<<256, 256, 0, stream>>>(sbuf, (long long)NN * HH, 0.f);

    // pre-pack operands to bf16 (halves GEMM read bytes; enables async b128)
    cvt_f32_bf16<<<2048, 256, 0, stream>>>(feat, featb, (long long)NF);
    cvt_f32_bf16<<<64, 256, 0, stream>>>(W_src, Wb, (long long)HD * IND);

    // 1) feat_src = feat @ W^T  (bf16 WMMA, f32 accumulate)
    dim3 ggrid((NN + 63) / 64, HD / 64);
    gemm_feat_wmma<<<ggrid, 128, 0, stream>>>(featb, Wb, h0);

    // h3 no longer needed as featb -> zero it for hop-3 accumulation
    fill_f32<<<2048, 256, 0, stream>>>(h3, (long long)NF, 0.f);

    // 2) attention logits per node
    el_er_kernel<<<(NN * 32 + 255) / 256, 256, 0, stream>>>(h0, attn_l, attn_r, el, er);

    // 3-5) edge softmax over dst
    int eblk = (EE + 255) / 256;
    edge_logit_max<<<eblk, 256, 0, stream>>>(el, er, src, dst, abuf, mbuf);
    edge_exp_sum<<<eblk, 256, 0, stream>>>(dst, abuf, mbuf, sbuf);
    edge_norm<<<eblk, 256, 0, stream>>>(dst, abuf, sbuf);

    // 6) K=3 diffusion hops (L2-resident: h_prev+h_next+a ~ 115MB < 192MB L2)
    spmm_hop<<<4096, 256, 0, stream>>>(h0, abuf, src, dst, h1);
    spmm_hop<<<4096, 256, 0, stream>>>(h1, abuf, src, dst, h2);
    spmm_hop<<<4096, 256, 0, stream>>>(h2, abuf, src, dst, h3);

    // 7) hop-wise attention + bias -> output [N,H,D]
    hop_attn_kernel<<<(NN * 32 + 255) / 256, 256, 0, stream>>>(
        h0, h1, h2, h3, hop_l, hop_r, pos, bias, out);
}
